// GearNetFromCoordinates_48936857370928
// MI455X (gfx1250) — matrix operand order repaired and checked by reading, not compiled, not run on hardware
//
#include <hip/hip_runtime.h>
#include <hip/hip_bf16.h>

#define BB 8
#define SS 2048
#define HH 512
#define RR 7
#define KK 10
#define LL 4
#define RH (RR*HH)        /* 3584 */
#define CAP 192           /* max reverse-KNN in-degree we record */
#define PAD 4             /* guard rows each side of hb per graph */
#define SP (SS + 2*PAD)   /* padded rows per graph: 2056 */

typedef __attribute__((ext_vector_type(16))) __bf16 v16bf;
typedef __attribute__((ext_vector_type(8)))  __bf16 v8bf;
typedef __attribute__((ext_vector_type(8)))  float  v8f;

// ---------------------------------------------------------------------------
// h0 = ca @ Wp.T + bp   (one thread per output element)
// ---------------------------------------------------------------------------
__global__ __launch_bounds__(256)
void gn_init_h(const float* __restrict__ ca, const float* __restrict__ Wp,
               const float* __restrict__ bp, float* __restrict__ h) {
  int t = blockIdx.x * 256 + threadIdx.x;
  if (t >= BB * SS * HH) return;
  int n  = t & (HH - 1);
  int gs = t / HH;                       // g*SS + s
  const float* p = ca + (size_t)gs * 3;
  h[t] = p[0] * Wp[n * 3 + 0] + p[1] * Wp[n * 3 + 1] + p[2] * Wp[n * 3 + 2] + bp[n];
}

// ---------------------------------------------------------------------------
// weight conversion fp32 -> bf16 (layouts unchanged: (L,H,R*H) and (L,H,H))
// ---------------------------------------------------------------------------
__global__ __launch_bounds__(256)
void gn_cvt_wlin(const float* __restrict__ w, __bf16* __restrict__ o) {
  int t = blockIdx.x * 256 + threadIdx.x;
  if (t < LL * HH * RH) o[t] = (__bf16)w[t];
}
__global__ __launch_bounds__(256)
void gn_cvt_wself(const float* __restrict__ w, __bf16* __restrict__ o) {
  int t = blockIdx.x * 256 + threadIdx.x;
  if (t < LL * HH * HH) o[t] = (__bf16)w[t];
}

// ---------------------------------------------------------------------------
// fold BN + biases:  s = gamma*rsqrt(var+eps),  b2 = (blin+bself-mean)*s + beta
// ---------------------------------------------------------------------------
__global__ __launch_bounds__(256)
void gn_prep_sb(const float* __restrict__ gamma, const float* __restrict__ beta,
                const float* __restrict__ mean,  const float* __restrict__ var,
                const float* __restrict__ blin,  const float* __restrict__ bself,
                float* __restrict__ sbuf, float* __restrict__ b2buf) {
  int t = blockIdx.x * 256 + threadIdx.x;
  if (t >= LL * HH) return;
  float sc = gamma[t] * rsqrtf(var[t] + 1e-5f);
  sbuf[t]  = sc;
  b2buf[t] = (blin[t] + bself[t] - mean[t]) * sc + beta[t];
}

// ---------------------------------------------------------------------------
// KNN: one block per (node j, graph g). d2 row in LDS, 10 argmin passes.
// ---------------------------------------------------------------------------
__global__ __launch_bounds__(256)
void gn_knn(const float* __restrict__ ca, int* __restrict__ knn) {
  __shared__ float d2[SS];
  __shared__ float rv[256];
  __shared__ int   ri[256];
  const int j = blockIdx.x, g = blockIdx.y, t = threadIdx.x;
  const float* pg = ca + (size_t)g * SS * 3;
  const float xj = pg[j * 3 + 0], yj = pg[j * 3 + 1], zj = pg[j * 3 + 2];
  for (int m = t; m < SS; m += 256) {
    float dx = pg[m * 3 + 0] - xj;
    float dy = pg[m * 3 + 1] - yj;
    float dz = pg[m * 3 + 2] - zj;
    d2[m] = (m == j) ? 3.4e38f : (dx * dx + dy * dy + dz * dz);
  }
  __syncthreads();
  for (int k = 0; k < KK; ++k) {
    float best = 3.4e38f; int bi = SS;
    for (int m = t; m < SS; m += 256) {
      float d = d2[m];
      if (d < best) { best = d; bi = m; }
    }
    rv[t] = best; ri[t] = bi;
    __syncthreads();
    for (int s = 128; s > 0; s >>= 1) {
      if (t < s) {
        float ov = rv[t + s]; int oi = ri[t + s];
        if (ov < rv[t] || (ov == rv[t] && oi < ri[t])) { rv[t] = ov; ri[t] = oi; }
      }
      __syncthreads();
    }
    if (t == 0) {
      knn[((size_t)g * SS + j) * KK + k] = ri[0];
      d2[ri[0]] = 3.4e38f;
    }
    __syncthreads();
  }
}

// ---------------------------------------------------------------------------
// Deterministic reverse adjacency: thread (g,j) scans all knn lists in
// ascending i order -> sorted list, no atomics, replay-deterministic sums.
// ---------------------------------------------------------------------------
__global__ __launch_bounds__(256)
void gn_build_rev(const int* __restrict__ knn, int* __restrict__ cnt,
                  int* __restrict__ lists) {
  int t = blockIdx.x * 256 + threadIdx.x;
  if (t >= BB * SS) return;
  int g = t / SS, j = t - g * SS;
  const int* kg = knn + (size_t)g * SS * KK;
  int* lst = lists + (size_t)t * CAP;
  int n = 0;
  for (int i = 0; i < SS; ++i) {
    #pragma unroll
    for (int k = 0; k < KK; ++k) {
      if (kg[i * KK + k] == j && n < CAP) lst[n++] = i;
    }
  }
  cnt[t] = n;
}

// ---------------------------------------------------------------------------
// knn-sum (relation 6 aggregation), fp32 accumulate -> bf16, per layer
// ---------------------------------------------------------------------------
__global__ __launch_bounds__(256)
void gn_ksum(const float* __restrict__ h, const int* __restrict__ cnt,
             const int* __restrict__ lists, __bf16* __restrict__ ksb) {
  const int j = blockIdx.x, g = blockIdx.y, t = threadIdx.x;
  const int idx = g * SS + j;
  const int n = cnt[idx];
  const int* lst = lists + (size_t)idx * CAP;
  const float* hg = h + (size_t)g * SS * HH;
  float a0 = 0.f, a1 = 0.f;
  for (int e = 0; e < n; ++e) {
    int i = lst[e];
    a0 += hg[(size_t)i * HH + t];
    a1 += hg[(size_t)i * HH + t + 256];
  }
  __bf16* kg = ksb + (size_t)idx * HH;
  kg[t] = (__bf16)a0;
  kg[t + 256] = (__bf16)a1;
}

// ---------------------------------------------------------------------------
// h (fp32, unpadded) -> hb (bf16, PAD guard rows per graph, interior)
// ---------------------------------------------------------------------------
__global__ __launch_bounds__(256)
void gn_cvt_h(const float* __restrict__ h, __bf16* __restrict__ hb) {
  int t = blockIdx.x * 256 + threadIdx.x;
  if (t >= BB * SS * HH) return;
  int col = t & (HH - 1);
  int gs  = t / HH;
  int g   = gs / SS;
  int s   = gs - g * SS;
  hb[((size_t)g * SP + s + PAD) * HH + col] = (__bf16)h[t];
}

// zero the guard rows of hb (once per call)
__global__ __launch_bounds__(256)
void gn_zero_hpad(__bf16* __restrict__ hb) {
  int t = blockIdx.x * 256 + threadIdx.x;
  if (t >= BB * 2 * PAD * HH) return;
  int col  = t & (HH - 1);
  int gp   = t / HH;                 // g * 2*PAD + p
  int g    = gp / (2 * PAD);
  int p    = gp - g * 2 * PAD;       // 0..7
  int prow = (p < PAD) ? p : (SS + PAD + (p - PAD));
  hb[((size_t)g * SP + prow) * HH + col] = (__bf16)0.f;
}

// ---------------------------------------------------------------------------
// Fused layer GEMM:  out(2048x512) = A(2048x4096) @ W(4096x512)
//   A K-chunks r=0..5: shifted h (guard rows are zero), r=6: knn-sum, r=7: h
//   W K-chunks r=0..6: Wlin[l] rows (natural (N,K) layout), r=7: Wself[l]
// Epilogue: y = acc*s[n] + b2[n]; relu; h += y  (in place on d_out)
// Block: 256 threads = 8 waves (4 M x 2 N), tile 128x128, wave = 2x4 WMMA.
// All fragment loads are unconditional global_load_b128 (no predication).
// ---------------------------------------------------------------------------
__global__ __launch_bounds__(256)
void gn_gemm(const __bf16* __restrict__ hb, const __bf16* __restrict__ ksb,
             const __bf16* __restrict__ WlinL, const __bf16* __restrict__ WselfL,
             const float* __restrict__ sbuf, const float* __restrict__ b2buf,
             float* __restrict__ h) {
  const int g    = blockIdx.z;
  const int lane = threadIdx.x & 31;
  const int wave = threadIdx.x >> 5;
  const int wm   = wave & 3;
  const int wn   = wave >> 2;
  const int m0   = blockIdx.x * 128 + wm * 32;   // 32 rows per wave
  const int n0   = blockIdx.y * 128 + wn * 64;   // 64 cols per wave
  const int lm   = lane & 15;
  const int lh   = lane >> 4;
  const int acb  = lh << 4;   // A fragment chunk base (K)
  const int bkb  = lh << 4;   // B fragment chunk base (K)

  // hbg points at logical row 0 of this graph's padded activations
  const __bf16* hbg = hb  + ((size_t)g * SP + PAD) * HH;
  const __bf16* ksg = ksb + (size_t)g * SS * HH;

  v8f acc[2][4] = {};

  #pragma unroll 1
  for (int r = 0; r < 8; ++r) {
    // A-source row pointers for the two M sub-tiles (guard rows are zero)
    const __bf16* abase[2];
    #pragma unroll
    for (int i = 0; i < 2; ++i) {
      int row = m0 + i * 16 + lm;
      if (r < 6) {
        int off = (r < 3) ? (r - 3) : (r - 2);   // {-3,-2,-1,1,2,3}
        abase[i] = hbg + (size_t)(row - off) * HH;   // row-off in [-3, S+2]
      } else if (r == 6) {
        abase[i] = ksg + (size_t)row * HH;
      } else {
        abase[i] = hbg + (size_t)row * HH;
      }
    }
    // B column pointers for the four N sub-tiles
    const __bf16* bbase[4];
    #pragma unroll
    for (int j = 0; j < 4; ++j) {
      int n = n0 + j * 16 + lm;
      bbase[j] = (r < 7) ? (WlinL + (size_t)n * RH + (size_t)r * HH)
                         : (WselfL + (size_t)n * HH);
    }
    #pragma unroll 2
    for (int c = 0; c < HH; c += 32) {
      v16bf af[2];
      #pragma unroll
      for (int i = 0; i < 2; ++i) {
        v8bf lo = *(const v8bf*)(abase[i] + c + acb);
        v8bf hi = *(const v8bf*)(abase[i] + c + acb + 8);
        af[i] = __builtin_shufflevector(lo, hi, 0, 1, 2, 3, 4, 5, 6, 7,
                                                8, 9, 10, 11, 12, 13, 14, 15);
      }
      v16bf bf[4];
      #pragma unroll
      for (int j = 0; j < 4; ++j) {
        v8bf lo = *(const v8bf*)(bbase[j] + c + bkb);
        v8bf hi = *(const v8bf*)(bbase[j] + c + bkb + 8);
        bf[j] = __builtin_shufflevector(lo, hi, 0, 1, 2, 3, 4, 5, 6, 7,
                                                8, 9, 10, 11, 12, 13, 14, 15);
      }
      #pragma unroll
      for (int i = 0; i < 2; ++i)
        #pragma unroll
        for (int j = 0; j < 4; ++j)
          acc[i][j] = __builtin_amdgcn_wmma_f32_16x16x32_bf16(
              false, af[i], false, bf[j], (short)0, acc[i][j], false, false);
    }
  }

  // epilogue: BN-fold + relu + residual, in place on h (this block's region)
  float* hg = h + (size_t)g * SS * HH;
  #pragma unroll
  for (int j = 0; j < 4; ++j) {
    int n = n0 + j * 16 + lm;
    float sc = sbuf[n], bb = b2buf[n];
    #pragma unroll
    for (int i = 0; i < 2; ++i) {
      #pragma unroll
      for (int e = 0; e < 8; ++e) {
        int row = m0 + i * 16 + (lh << 3) + e;
        float y = acc[i][j][e] * sc + bb;
        y = y > 0.f ? y : 0.f;
        size_t o = (size_t)row * HH + n;
        hg[o] = y + hg[o];
      }
    }
  }
}

// ---------------------------------------------------------------------------
extern "C" void kernel_launch(void* const* d_in, const int* in_sizes, int n_in,
                              void* d_out, int out_size, void* d_ws, size_t ws_size,
                              hipStream_t stream) {
  const float* ca    = (const float*)d_in[1];
  const float* Wp    = (const float*)d_in[3];
  const float* bp    = (const float*)d_in[4];
  const float* Wlin  = (const float*)d_in[5];
  const float* blin  = (const float*)d_in[6];
  const float* Wself = (const float*)d_in[7];
  const float* bself = (const float*)d_in[8];
  const float* gamma = (const float*)d_in[9];
  const float* beta  = (const float*)d_in[10];
  const float* mean  = (const float*)d_in[11];
  const float* var   = (const float*)d_in[12];
  float* h = (float*)d_out;              // running hidden state, fp32

  // workspace carve (all 256B aligned)
  char* ws = (char*)d_ws;
  size_t o = 0;
  auto carve = [&](size_t bytes) {
    void* p = ws + o;
    o = (o + bytes + 255) & ~(size_t)255;
    return p;
  };
  __bf16* WlinB  = (__bf16*)carve((size_t)LL * HH * RH * 2);   // 14.7 MB
  __bf16* WselfB = (__bf16*)carve((size_t)LL * HH * HH * 2);   //  2.1 MB
  float*  sbuf   = (float*) carve((size_t)LL * HH * 4);
  float*  b2buf  = (float*) carve((size_t)LL * HH * 4);
  __bf16* hb     = (__bf16*)carve((size_t)BB * SP * HH * 2);   // 16.8 MB (padded)
  __bf16* ksb    = (__bf16*)carve((size_t)BB * SS * HH * 2);   // 16.8 MB
  int*    knn    = (int*)   carve((size_t)BB * SS * KK * 4);
  int*    cnt    = (int*)   carve((size_t)BB * SS * 4);
  int*    lists  = (int*)   carve((size_t)BB * SS * CAP * 4);  // 12.6 MB

  const int NE = BB * SS * HH;           // 8,388,608

  gn_cvt_wlin <<<(LL * HH * RH + 255) / 256, 256, 0, stream>>>(Wlin, WlinB);
  gn_cvt_wself<<<(LL * HH * HH + 255) / 256, 256, 0, stream>>>(Wself, WselfB);
  gn_prep_sb  <<<(LL * HH + 255) / 256, 256, 0, stream>>>(gamma, beta, mean, var,
                                                          blin, bself, sbuf, b2buf);
  gn_init_h   <<<(NE + 255) / 256, 256, 0, stream>>>(ca, Wp, bp, h);
  gn_zero_hpad<<<(BB * 2 * PAD * HH + 255) / 256, 256, 0, stream>>>(hb);
  gn_knn      <<<dim3(SS, BB), 256, 0, stream>>>(ca, knn);
  gn_build_rev<<<(BB * SS + 255) / 256, 256, 0, stream>>>(knn, cnt, lists);

  for (int l = 0; l < LL; ++l) {
    gn_ksum  <<<dim3(SS, BB), 256, 0, stream>>>(h, cnt, lists, ksb);
    gn_cvt_h <<<(NE + 255) / 256, 256, 0, stream>>>(h, hb);
    gn_gemm  <<<dim3(SS / 128, HH / 128, BB), 256, 0, stream>>>(
        hb, ksb,
        WlinB  + (size_t)l * HH * RH,
        WselfB + (size_t)l * HH * HH,
        sbuf + l * HH, b2buf + l * HH, h);
  }
}
